// branch_1_module_82411832475692
// MI455X (gfx1250) — matrix operand order repaired and checked by reading, not compiled, not run on hardware
//
#include <hip/hip_runtime.h>

// ---------------------------------------------------------------------------
// CDNA5 / gfx1250 types
// ---------------------------------------------------------------------------
typedef __attribute__((ext_vector_type(16))) __bf16 v16bf;
typedef __attribute__((ext_vector_type(8)))  float  v8f;
typedef __attribute__((ext_vector_type(4))) unsigned int u32x4;
typedef __attribute__((ext_vector_type(8)))  int    i32x8;
typedef __attribute__((ext_vector_type(4)))  int    i32x4;

enum { OP_NONE = 0, OP_RELU = 1, OP_LEAKY = 2, OP_SIGN = 3 };
enum { CF_RELU = 1, CF_ADDRES = 2 };

#if defined(__HIP_DEVICE_COMPILE__) && __has_builtin(__builtin_amdgcn_tensor_load_to_lds)
#define ATHENA_TDM 1
#endif

#ifdef ATHENA_TDM
// Issue (no wait) a TDM DMA of a [tileD1 rows x tileD0 cols] f32 tile
// (row stride = strideElems) from global into LDS at ldsAddr.
// Tensor dims give free OOB zero-fill for K / M tails.
__device__ __forceinline__ void tdm_issue_2d_f32(unsigned ldsAddr, const float* gptr,
                                                 unsigned tensorD0, unsigned tensorD1,
                                                 unsigned tileD0, unsigned tileD1,
                                                 unsigned long long strideElems)
{
  unsigned long long ga = (unsigned long long)(const void*)gptr;
  u32x4 g0;
  g0[0] = 1u;                                     // count=1, user mode, no gather
  g0[1] = ldsAddr;                                // LDS byte address
  g0[2] = (unsigned)(ga & 0xffffffffu);           // global_addr[31:0]
  g0[3] = (unsigned)((ga >> 32) & 0x01ffffffu) | (2u << 30);  // addr[56:32] | type=2
  i32x8 g1;
  g1[0] = (int)(2u << 16);                        // wg_mask=0, data_size=4B
  g1[1] = (int)((tensorD0 & 0xffffu) << 16);      // tensor_dim0[15:0]
  g1[2] = (int)((tensorD0 >> 16) | ((tensorD1 & 0xffffu) << 16));
  g1[3] = (int)((tensorD1 >> 16) | (tileD0 << 16));
  g1[4] = (int)(tileD1 & 0xffffu);                // tile_dim1, tile_dim2=0
  g1[5] = (int)(strideElems & 0xffffffffu);       // tensor_dim0_stride lo
  g1[6] = (int)((strideElems >> 32) & 0xffffu);   // stride0 hi, stride1=0
  g1[7] = 0;
  i32x4 z4 = {0, 0, 0, 0};
#if __clang_major__ >= 23
  i32x8 z8 = {0, 0, 0, 0, 0, 0, 0, 0};
  __builtin_amdgcn_tensor_load_to_lds(g0, g1, z4, z4, z8, 0);
#else
  __builtin_amdgcn_tensor_load_to_lds(g0, g1, z4, z4, 0);
#endif
}
#endif

// ---------------------------------------------------------------------------
// Implicit-GEMM conv + folded BN (+residual, +relu) with bf16 WMMA.
// Block = 256 thr = 8 waves; block tile = 16 CO x 512 pixels.
// Each wave owns 16 CO x 64 pixels = 4 accumulator tiles -> 4 WMMAs / K-step.
// Weight panel double-buffered via TDM: DMA for step s+1 is issued before
// waiting (tensorcnt<=1) so the tensor engine overlaps the WMMA work.
// ---------------------------------------------------------------------------
__global__ void __launch_bounds__(256)
conv_wmma_kernel(const float* __restrict__ in, const float* __restrict__ wgt,
                 const float* __restrict__ bnsc, const float* __restrict__ bnbi,
                 const float* __restrict__ res, float* __restrict__ out,
                 int B, int CI, int IH, int IW, int CO, int OH, int OW,
                 int KH, int KW, int stride, int pad, int flags)
{
  __shared__ float aTile[2][16 * 32];            // double-buffered f32 weight panel
  __shared__ int sOff[32];                       // ci*IH*IW or -1 (k OOB)
  __shared__ int sKhk[32];                       // (kh<<8) | kw
  const int Ktot = CI * KH * KW;
  const int khw  = KH * KW;
  const int P    = B * OH * OW;
  const int wave = threadIdx.x >> 5;
  const int lane = threadIdx.x & 31;
  const int coBase = blockIdx.y * 16;
  const int gBase  = blockIdx.x * 512 + wave * 64;

  const int baseA = (lane < 16) ? 0 : 8;
  const int baseB = (lane < 16) ? 0 : 16;
  const int m16   = lane & 15;

  // 4 output pixel tiles per wave
  int  gv[4], pbv[4], pyv[4], pxv[4], pys[4], pxs[4];
  long ibase[4];
#pragma unroll
  for (int t = 0; t < 4; ++t) {
    int g = gBase + t * 16 + m16;
    gv[t] = g;
    int pb = 0, py = 0, px = 0;
    if (g < P) {
      pb = g / (OH * OW);
      int r = g - pb * OH * OW;
      py = r / OW;
      px = r - py * OW;
    }
    pbv[t] = pb; pyv[t] = py; pxv[t] = px;
    pys[t] = py * stride - pad;
    pxs[t] = px * stride - pad;
    ibase[t] = (long)pb * CI * IH * IW;
  }
  if (gv[0] < P) __builtin_prefetch(in + ibase[0], 0, 1);

  const int nsteps = (Ktot + 31) >> 5;
#ifdef ATHENA_TDM
  if (wave == 0) {
    tdm_issue_2d_f32((unsigned)(size_t)(void*)aTile[0],
                     wgt + (size_t)coBase * Ktot,
                     (unsigned)Ktot, (unsigned)(CO - coBase),
                     32u, 16u, (unsigned long long)Ktot);
  }
#endif

  v8f acc[4] = {};
  for (int s = 0; s < nsteps; ++s) {
    const int k0 = s << 5;
#ifdef ATHENA_TDM
    if (wave == 0) {
      if (s + 1 < nsteps) {
        tdm_issue_2d_f32((unsigned)(size_t)(void*)aTile[(s + 1) & 1],
                         wgt + (size_t)coBase * Ktot + (k0 + 32),
                         (unsigned)(Ktot - (k0 + 32)), (unsigned)(CO - coBase),
                         32u, 16u, (unsigned long long)Ktot);
        __builtin_amdgcn_s_wait_tensorcnt((short)1);   // step s done, s+1 in flight
      } else {
        __builtin_amdgcn_s_wait_tensorcnt((short)0);
      }
      // k -> (ci, kh, kw) decode table for this K-step
      int kk = k0 + lane;
      if (kk < Ktot) {
        int ci = kk / khw;
        int r2 = kk - ci * khw;
        int kh = r2 / KW;
        int kw = r2 - kh * KW;
        sOff[lane] = ci * IH * IW;
        sKhk[lane] = (kh << 8) | kw;
      } else {
        sOff[lane] = -1;
        sKhk[lane] = 0;
      }
    }
#else
    for (int idx = threadIdx.x; idx < 512; idx += 256) {
      int rr = idx >> 5, cc = idx & 31;
      int co = coBase + rr, kk = k0 + cc;
      aTile[0][idx] = (co < CO && kk < Ktot) ? wgt[(long)co * Ktot + kk] : 0.f;
    }
    if (threadIdx.x < 32) {
      int kk = k0 + (int)threadIdx.x;
      if (kk < Ktot) {
        int ci = kk / khw;
        int r2 = kk - ci * khw;
        int kh = r2 / KW;
        int kw = r2 - kh * KW;
        sOff[threadIdx.x] = ci * IH * IW;
        sKhk[threadIdx.x] = (kh << 8) | kw;
      } else {
        sOff[threadIdx.x] = -1;
        sKhk[threadIdx.x] = 0;
      }
    }
#endif
    __syncthreads();

#ifdef ATHENA_TDM
    const float* aCur = aTile[s & 1];
#else
    const float* aCur = aTile[0];
#endif
    // A fragment (weights) from LDS -- shared by all 4 WMMAs
    v16bf af;
#pragma unroll
    for (int j = 0; j < 16; ++j) {
      int kk = baseB + j;
      af[j] = (__bf16)aCur[m16 * 32 + kk];
    }
    // 4 B fragments / 4 WMMAs
#pragma unroll
    for (int t = 0; t < 4; ++t) {
      v16bf bfv;
#pragma unroll
      for (int j = 0; j < 16; ++j) {
        int kl = baseB + j;
        int off = sOff[kl];
        float v = 0.f;
        if (gv[t] < P && off >= 0) {
          int kh = sKhk[kl] >> 8;
          int kw = sKhk[kl] & 255;
          int iy = pys[t] + kh;
          int ix = pxs[t] + kw;
          if (iy >= 0 && iy < IH && ix >= 0 && ix < IW)
            v = in[ibase[t] + off + (long)iy * IW + ix];
        }
        bfv[j] = (__bf16)v;
      }
      acc[t] = __builtin_amdgcn_wmma_f32_16x16x32_bf16(false, af, false, bfv,
                                                       (short)0, acc[t], false, false);
    }
    __syncthreads();
  }

  const int mAdd = (lane < 16) ? 0 : 8;
#pragma unroll
  for (int t = 0; t < 4; ++t) {
#pragma unroll
    for (int r = 0; r < 8; ++r) {
      int co = coBase + r + mAdd;
      if (co < CO && gv[t] < P) {
        float v = acc[t][r] * bnsc[co] + bnbi[co];
        long oidx = ((long)(pbv[t] * CO + co) * OH + pyv[t]) * OW + pxv[t];
        if (flags & CF_ADDRES) v += res[oidx];
        if (flags & CF_RELU) v = fmaxf(v, 0.f);
        out[oidx] = v;
      }
    }
  }
}

// ---------------------------------------------------------------------------
// Generic batched bf16-WMMA GEMM: C = A(MxK,rm) x B (+bias[n]) with epilogue.
// Block = 128 thr = 4 waves; wave tile 16x32 (2 acc tiles), K unrolled by 2
// -> 4 WMMAs per iteration. A panel [16x64] double-buffered in LDS via TDM
// (shared by all waves in the block). Masked-zero loads legalize any K.
// ---------------------------------------------------------------------------
__global__ void __launch_bounds__(128)
gemm_wmma_kernel(const float* __restrict__ A, const float* __restrict__ Bm,
                 const float* __restrict__ bias, float* __restrict__ C,
                 int M, int N, int K, int transB,
                 long sA, long sB, long sC, int op)
{
  __shared__ float aTile[2][16 * 64];
  const int wave = threadIdx.x >> 5;
  const int lane = threadIdx.x & 31;
  const int mBase = blockIdx.y * 16;
  const int nBase = (blockIdx.x * 8 + wave * 2) * 16;   // block covers 128 cols
  const float* Ab = A + (long)blockIdx.z * sA;
  const float* Bb = Bm + (long)blockIdx.z * sB;
  float* Cb = C + (long)blockIdx.z * sC;

  const int m16   = lane & 15;
  const int baseA = (lane < 16) ? 0 : 8;
  const int baseB = (lane < 16) ? 0 : 16;
  const int nCol[2] = { nBase + m16, nBase + 16 + m16 };

  __builtin_prefetch(Bb, 0, 1);

  const int nsteps = (K + 63) >> 6;
#ifdef ATHENA_TDM
  if (wave == 0) {
    tdm_issue_2d_f32((unsigned)(size_t)(void*)aTile[0],
                     Ab + (long)mBase * K,
                     (unsigned)K, (unsigned)(M - mBase),
                     64u, 16u, (unsigned long long)K);
  }
#endif

  v8f acc[2] = {};
  for (int s = 0; s < nsteps; ++s) {
    const int k0 = s << 6;
#ifdef ATHENA_TDM
    if (wave == 0) {
      if (s + 1 < nsteps) {
        tdm_issue_2d_f32((unsigned)(size_t)(void*)aTile[(s + 1) & 1],
                         Ab + (long)mBase * K + (k0 + 64),
                         (unsigned)(K - (k0 + 64)), (unsigned)(M - mBase),
                         64u, 16u, (unsigned long long)K);
        __builtin_amdgcn_s_wait_tensorcnt((short)1);   // overlap next DMA with WMMA
      } else {
        __builtin_amdgcn_s_wait_tensorcnt((short)0);
      }
    }
#else
    for (int idx = threadIdx.x; idx < 1024; idx += 128) {
      int rr = idx >> 6, cc = idx & 63;
      int mm = mBase + rr, kk = k0 + cc;
      aTile[0][idx] = (mm < M && kk < K) ? Ab[(long)mm * K + kk] : 0.f;
    }
#endif
    __syncthreads();

#ifdef ATHENA_TDM
    const float* aCur = aTile[s & 1];
#else
    const float* aCur = aTile[0];
#endif
#pragma unroll
    for (int half = 0; half < 2; ++half) {
      const int kh0 = k0 + half * 32;
      v16bf af;
#pragma unroll
      for (int j = 0; j < 16; ++j) {
        int kk = baseB + j;
        af[j] = (__bf16)aCur[m16 * 64 + half * 32 + kk];
      }
#pragma unroll
      for (int t = 0; t < 2; ++t) {
        v16bf bfv;
#pragma unroll
        for (int j = 0; j < 16; ++j) {
          int kk = kh0 + baseB + j;
          float v = 0.f;
          if (nCol[t] < N && kk < K)
            v = transB ? Bb[(long)nCol[t] * K + kk] : Bb[(long)kk * N + nCol[t]];
          bfv[j] = (__bf16)v;
        }
        acc[t] = __builtin_amdgcn_wmma_f32_16x16x32_bf16(false, af, false, bfv,
                                                         (short)0, acc[t], false, false);
      }
    }
    __syncthreads();
  }

  const int mAdd = (lane < 16) ? 0 : 8;
#pragma unroll
  for (int t = 0; t < 2; ++t) {
#pragma unroll
    for (int r = 0; r < 8; ++r) {
      int m = mBase + r + mAdd;
      if (m < M && nCol[t] < N) {
        float v = acc[t][r];
        if (bias) v += bias[nCol[t]];
        if (op == OP_RELU)       v = fmaxf(v, 0.f);
        else if (op == OP_LEAKY) v = (v > 0.f) ? v : 0.05f * v;
        else if (op == OP_SIGN)  v = (v > 0.f) ? 1.f : ((v < 0.f) ? -1.f : 0.f);
        Cb[(long)m * N + nCol[t]] = v;
      }
    }
  }
}

// ---------------------------------------------------------------------------
// Small helper kernels
// ---------------------------------------------------------------------------
__global__ void patchify_kernel(const float* __restrict__ x, float* __restrict__ xp)
{
  long i = (long)blockIdx.x * 256 + threadIdx.x;
  const long total = 512L * 3 * 64 * 64;
  if (i >= total) return;
  int w = (int)(i & 63);
  long t = i >> 6;
  int h = (int)(t & 63); t >>= 6;
  int c = (int)(t % 3);  t /= 3;
  int patch = (int)t;
  int q = patch & 7;
  int p = (patch >> 3) & 7;
  int nn = patch >> 6;
  xp[i] = x[(((long)nn * 3 + c) * 512 + (p * 64 + h)) * 512 + (q * 64 + w)];
}

__global__ void bn_prep_kernel(const float* __restrict__ s, const float* __restrict__ b,
                               const float* __restrict__ m, const float* __restrict__ v,
                               float* __restrict__ sc, float* __restrict__ bi, int C)
{
  int i = blockIdx.x * 64 + threadIdx.x;
  if (i < C) {
    float k = s[i] * rsqrtf(v[i] + 1e-5f);
    sc[i] = k;
    bi[i] = b[i] - m[i] * k;
  }
}

__global__ void maxpool_kernel(const float* __restrict__ in, float* __restrict__ out,
                               long BC, int IH, int IW, int OH, int OW)
{
  long i = (long)blockIdx.x * 256 + threadIdx.x;
  long total = BC * OH * OW;
  if (i >= total) return;
  int ox = (int)(i % OW);
  long t = i / OW;
  int oy = (int)(t % OH);
  long bc = t / OH;
  float mv = -3.4e38f;
  for (int dy = -1; dy <= 1; ++dy)
    for (int dx = -1; dx <= 1; ++dx) {
      int iy = oy * 2 + dy, ix = ox * 2 + dx;
      if (iy >= 0 && iy < IH && ix >= 0 && ix < IW)
        mv = fmaxf(mv, in[(bc * IH + iy) * IW + ix]);
    }
  out[i] = mv;
}

__device__ __forceinline__ float block_reduce_sum(float v, float* red)
{
  int t = threadIdx.x;
  red[t] = v;
  __syncthreads();
  for (int s = blockDim.x >> 1; s > 0; s >>= 1) {
    if (t < s) red[t] += red[t + s];
    __syncthreads();
  }
  float r = red[0];
  __syncthreads();
  return r;
}

// att_cell: f [512,512,2,2] -> out_cell [512,512], beta [512,4], y_beta [512,4,2]
__global__ void __launch_bounds__(256)
att_cell_kernel(const float* __restrict__ f, const float* __restrict__ lw,
                const float* __restrict__ lb, float* __restrict__ out_cell,
                float* __restrict__ beta, float* __restrict__ ybeta)
{
  int b = blockIdx.x;
  __shared__ float red[256];
  __shared__ float anorm[4];
  __shared__ float dots[8];
  __shared__ float aw[4];
  for (int gq = 0; gq < 4; ++gq) {
    float p = 0.f;
    for (int d = threadIdx.x; d < 512; d += 256) {
      float v = f[((long)b * 512 + d) * 4 + gq];
      p += v * v;
    }
    float s = block_reduce_sum(p, red);
    if (threadIdx.x == 0) anorm[gq] = sqrtf(s);
    __syncthreads();
  }
  for (int gq = 0; gq < 4; ++gq)
    for (int cls = 0; cls < 2; ++cls) {
      float p = 0.f;
      for (int d = threadIdx.x; d < 512; d += 256)
        p += f[((long)b * 512 + d) * 4 + gq] * lw[cls * 512 + d];
      float s = block_reduce_sum(p, red);
      if (threadIdx.x == 0) dots[gq * 2 + cls] = s + lb[cls];
      __syncthreads();
    }
  if (threadIdx.x == 0) {
    float s1 = anorm[0] + anorm[1] + anorm[2] + anorm[3];
    float s2 = 0.f;
    for (int gq = 0; gq < 4; ++gq) {
      float t = fmaxf(anorm[gq] / s1 - 0.075f, 0.f);   // 0.3/gamma, gamma=4
      aw[gq] = t;
      s2 += t;
    }
    for (int gq = 0; gq < 4; ++gq) {
      aw[gq] /= s2;
      beta[b * 4 + gq] = aw[gq];
    }
    for (int q = 0; q < 8; ++q) ybeta[b * 8 + q] = dots[q];
  }
  __syncthreads();
  for (int d = threadIdx.x; d < 512; d += 256) {
    float v = 0.f;
    for (int gq = 0; gq < 4; ++gq)
      v += aw[gq] * f[((long)b * 512 + d) * 4 + gq];
    out_cell[(long)b * 512 + d] = v;
  }
}

// adjacency build per sample: Qin [8,64,64] -> Aout [8,64,64]
__global__ void __launch_bounds__(256)
adj_kernel(const float* __restrict__ Qin, float* __restrict__ Aout)
{
  int b = blockIdx.x;
  __shared__ float Q[4096];
  __shared__ float A2[4096];
  __shared__ float red[256];
  __shared__ float cn[64];
  __shared__ float rsum[64];
  __shared__ float meanv;
  int tid = threadIdx.x;
  const float* src = Qin + (long)b * 4096;
  for (int i = tid; i < 4096; i += 256) Q[i] = src[i];
  __syncthreads();
  float p = 0.f;
  for (int i = tid; i < 4096; i += 256) p += Q[i];
  float s = block_reduce_sum(p, red);
  if (tid == 0) meanv = s / 4096.f;
  __syncthreads();
  for (int i = tid; i < 4096; i += 256) Q[i] = fmaxf(Q[i] - meanv, 0.f);
  __syncthreads();
  if (tid < 64) {
    float cs = 0.f;
    for (int r = 0; r < 64; ++r) { float v = Q[r * 64 + tid]; cs += v * v; }
    cn[tid] = sqrtf(cs);
  }
  __syncthreads();
  for (int i = tid; i < 4096; i += 256) {
    int c = i & 63;
    Q[i] = Q[i] / fmaxf(cn[c], 1e-12f);
  }
  __syncthreads();
  if (tid < 64) Q[tid * 64 + tid] += 1e-7f;
  __syncthreads();
  for (int i = tid; i < 4096; i += 256) {
    int r = i >> 6, c = i & 63;
    A2[i] = 0.5f * (Q[i] + Q[c * 64 + r]);
  }
  __syncthreads();
  if (tid < 64) {
    float rs = 0.f;
    for (int c = 0; c < 64; ++c) rs += A2[tid * 64 + c];
    rsum[tid] = (rs > 0.f) ? rsqrtf(rs) : 0.f;
  }
  __syncthreads();
  for (int i = tid; i < 4096; i += 256) {
    int r = i >> 6, c = i & 63;
    Aout[(long)b * 4096 + i] = A2[i] * rsum[r] * rsum[c];   // A symmetric
  }
}

// att_patch: x [8,64,512] -> o [8,2], emb [8,512], alpha [8,64]
__global__ void __launch_bounds__(256)
att_patch_kernel(const float* __restrict__ x, const float* __restrict__ lw,
                 const float* __restrict__ lb, float* __restrict__ o,
                 float* __restrict__ emb, float* __restrict__ alpha)
{
  int b = blockIdx.x;
  __shared__ float red[256];
  __shared__ float aw[64];
  __shared__ float sOut[512];
  int tid = threadIdx.x;
  if (tid < 64) {
    float s = 0.f;
    for (int d = 0; d < 512; ++d) {
      float v = x[((long)b * 64 + tid) * 512 + d];
      s += v * v;
    }
    aw[tid] = sqrtf(s);
  }
  __syncthreads();
  if (tid == 0) {
    float s1 = 0.f;
    for (int pp = 0; pp < 64; ++pp) s1 += aw[pp];
    float s2 = 0.f;
    for (int pp = 0; pp < 64; ++pp) {
      float t = fmaxf(aw[pp] / s1 - 0.003125f, 0.f);   // 0.2/gamma, gamma=64
      aw[pp] = t;
      s2 += t;
    }
    for (int pp = 0; pp < 64; ++pp) {
      aw[pp] /= s2;
      alpha[b * 64 + pp] = aw[pp];
    }
  }
  __syncthreads();
  for (int d = tid; d < 512; d += 256) {
    float v = 0.f;
    for (int pp = 0; pp < 64; ++pp)
      v += aw[pp] * x[((long)b * 64 + pp) * 512 + d];
    sOut[d] = v;
    emb[b * 512 + d] = v;
  }
  __syncthreads();
  for (int cls = 0; cls < 2; ++cls) {
    float p = 0.f;
    for (int d = tid; d < 512; d += 256) p += sOut[d] * lw[cls * 512 + d];
    float s = block_reduce_sum(p, red);
    if (tid == 0) o[b * 2 + cls] = s + lb[cls];
    __syncthreads();
  }
}

// ---------------------------------------------------------------------------
// Host driver
// ---------------------------------------------------------------------------
struct Blk { int c1, b1, c2, b2, sc, sb, planes, stride; };
static const Blk g_blocks[8] = {
  { 8,  9, 13, 14, -1, -1,  64, 1},
  {18, 19, 23, 24, -1, -1,  64, 1},
  {28, 29, 33, 34, 38, 39, 128, 2},
  {43, 44, 48, 49, -1, -1, 128, 1},
  {53, 54, 58, 59, 63, 64, 256, 2},
  {68, 69, 73, 74, -1, -1, 256, 1},
  {78, 79, 83, 84, 88, 89, 512, 2},
  {93, 94, 98, 99, -1, -1, 512, 1},
};

extern "C" void kernel_launch(void* const* d_in, const int* in_sizes, int n_in,
                              void* d_out, int out_size, void* d_ws, size_t ws_size,
                              hipStream_t stream)
{
  (void)in_sizes; (void)n_in; (void)out_size; (void)ws_size;
#define IN(i) ((const float*)d_in[(i)])
  float* W = (float*)d_ws;
  float* dout = (float*)d_out;

  // workspace layout (floats)
  size_t off = 0;
  auto take = [&](size_t n) { size_t r = off; off += n; return r; };
  const size_t XP   = take(512ul * 3 * 64 * 64);
  const size_t C1   = take(512ul * 64 * 32 * 32);
  const size_t POOL = take(512ul * 64 * 16 * 16);
  const size_t ACT0 = take(512ul * 64 * 16 * 16);
  const size_t ACT1 = take(512ul * 64 * 16 * 16);
  const size_t TMPB = take(512ul * 64 * 16 * 16);
  const size_t SCB  = take(512ul * 64 * 16 * 16);
  const size_t BNSC = take(32ul * 512);
  const size_t BNBI = take(32ul * 512);
  const size_t OUTCELL = take(512ul * 512);
  const size_t HGCN = take(512ul * 512);
  const size_t MMAT = take(512ul * 128);
  const size_t ZL   = take(8ul * 64 * 128);
  const size_t QB   = take(8ul * 64 * 64);
  const size_t AADJ = take(8ul * 64 * 64);
  const size_t XW1  = take(8ul * 64 * 512);
  const size_t H1B  = take(8ul * 64 * 512);
  const size_t HW2  = take(8ul * 64 * 512);
  const size_t GOUT = take(8ul * 64 * 512);

  // d_out offsets (tuple order: o, emb, alpha, out_alpha, beta, y_beta, hc)
  const size_t O_O = 0, O_EMB = 16, O_ALPHA = 4112, O_OA = 4624;
  const size_t O_BETA = 266768, O_YBETA = 268816, O_HC = 272912;

  auto bnprep = [&](int idx, int slot, int C) {
    bn_prep_kernel<<<dim3((C + 63) / 64), dim3(64), 0, stream>>>(
        IN(idx), IN(idx + 1), IN(idx + 2), IN(idx + 3),
        W + BNSC + slot * 512, W + BNBI + slot * 512, C);
  };
  auto conv = [&](const float* in, int widx, int slot, const float* res, float* out,
                  int B, int CI, int IH, int CO, int OH, int KH, int st, int pad, int flags) {
    int P = B * OH * OH;
    dim3 grid((P + 511) / 512, (CO + 15) / 16, 1);
    conv_wmma_kernel<<<grid, dim3(256), 0, stream>>>(
        in, IN(widx), W + BNSC + slot * 512, W + BNBI + slot * 512, res, out,
        B, CI, IH, IH, CO, OH, OH, KH, KH, st, pad, flags);
  };
  auto gemm = [&](const float* A, const float* B, const float* bias, float* C,
                  int M, int N, int K, int transB, long sA, long sB, long sC,
                  int batch, int op) {
    dim3 grid((N + 127) / 128, (M + 15) / 16, batch);
    gemm_wmma_kernel<<<grid, dim3(128), 0, stream>>>(A, B, bias, C, M, N, K,
                                                     transB, sA, sB, sC, op);
  };

  // ---- patchify: [8,3,512,512] -> [512,3,64,64]
  {
    long total = 512L * 3 * 64 * 64;
    patchify_kernel<<<dim3((unsigned)((total + 255) / 256)), dim3(256), 0, stream>>>(IN(0), W + XP);
  }

  // ---- conv1 (7x7 s2 p3) + bn + relu; maxpool 3x3 s2 p1
  bnprep(4, 0, 64);
  conv(W + XP, 3, 0, nullptr, W + C1, 512, 3, 64, 64, 32, 7, 2, 3, CF_RELU);
  {
    long total = 512L * 64 * 16 * 16;
    maxpool_kernel<<<dim3((unsigned)((total + 255) / 256)), dim3(256), 0, stream>>>(
        W + C1, W + POOL, 512L * 64, 32, 32, 16, 16);
  }

  // ---- residual blocks
  int slot = 1;
  const float* cur = W + POOL;
  int curC = 64, curH = 16, flip = 0;
  float* actbuf[2] = {W + ACT0, W + ACT1};
  for (int bi = 0; bi < 8; ++bi) {
    const Blk& bk = g_blocks[bi];
    int st = bk.stride;
    int OH = curH / st;
    int s1 = slot++; bnprep(bk.b1, s1, bk.planes);
    conv(cur, bk.c1, s1, nullptr, W + TMPB, 512, curC, curH, bk.planes, OH, 3, st, 1, CF_RELU);
    const float* resp = cur;
    if (bk.sc >= 0) {
      int ss = slot++; bnprep(bk.sb, ss, bk.planes);
      conv(cur, bk.sc, ss, nullptr, W + SCB, 512, curC, curH, bk.planes, OH, 1, st, 0, 0);
      resp = W + SCB;
    }
    int s2 = slot++; bnprep(bk.b2, s2, bk.planes);
    float* nxt = actbuf[flip]; flip ^= 1;
    conv(W + TMPB, bk.c2, s2, resp, nxt, 512, bk.planes, OH, bk.planes, OH, 3, 1, 1,
         CF_RELU | CF_ADDRES);
    cur = nxt; curC = bk.planes; curH = OH;
  }
  // cur = f: [512, 512, 2, 2]

  // ---- att_cell -> out_cell [8,64,512], beta, y_beta
  att_cell_kernel<<<dim3(512), dim3(256), 0, stream>>>(
      cur, IN(103), IN(104), W + OUTCELL, dout + O_BETA, dout + O_YBETA);

  // ---- pre_gcn
  gemm(IN(107), IN(108), IN(109), W + HGCN, 512, 512, 512, 1, 0, 0, 0, 1, OP_RELU); // temp@V0w.T
  gemm(W + HGCN, IN(110), IN(111), W + MMAT, 512, 128, 512, 1, 0, 0, 0, 1, OP_NONE); // @V1w.T
  gemm(W + OUTCELL, W + MMAT, nullptr, W + ZL, 64, 128, 512, 0, 32768, 0, 8192, 8, OP_LEAKY); // L
  gemm(W + ZL, W + ZL, nullptr, W + QB, 64, 64, 128, 1, 8192, 8192, 4096, 8, OP_NONE);        // Q=L@L.T
  adj_kernel<<<dim3(8), dim3(256), 0, stream>>>(W + QB, W + AADJ);
  gemm(W + OUTCELL, IN(112), nullptr, W + XW1, 64, 512, 512, 0, 32768, 0, 32768, 8, OP_NONE); // x@W1
  gemm(W + AADJ, W + XW1, IN(113), W + H1B, 64, 512, 64, 0, 4096, 32768, 32768, 8, OP_RELU);  // A@.+b1
  gemm(W + H1B, IN(114), nullptr, W + HW2, 64, 512, 512, 0, 32768, 0, 32768, 8, OP_NONE);     // h1@W2
  gemm(W + AADJ, W + HW2, IN(115), W + GOUT, 64, 512, 64, 0, 4096, 32768, 32768, 8, OP_NONE); // A@.+b2

  // ---- att_patch
  gemm(W + GOUT, IN(105), IN(106), dout + O_HC, 64, 64, 512, 1, 32768, 0, 4096, 8, OP_SIGN);  // hc
  gemm(dout + O_HC, dout + O_HC, nullptr, dout + O_OA, 512, 512, 64, 1, 0, 0, 0, 1, OP_NONE); // out_alpha
  att_patch_kernel<<<dim3(8), dim3(256), 0, stream>>>(
      W + GOUT, IN(103), IN(104), dout + O_O, dout + O_EMB, dout + O_ALPHA);
#undef IN
}
